// INT4Linear_19224273616899
// MI455X (gfx1250) — compile-verified, matrix operand order checked by reading, not run on hardware
//
#include <hip/hip_runtime.h>
#include <hip/hip_bf16.h>

// INT4 grouped-quant linear: Y[M,N] = X[M,K] * dequant(W[N,K])^T + bias
// M=8192, K=4096, N=12288, group=32 along K.
// bf16 WMMA (v_wmma_f32_16x16x32_bf16), f32 accumulate, double-buffered LDS.
// Dequant folded: (q-8)*s + z == q*s + (z - 8*s), zadj computed once per tile.

typedef __attribute__((ext_vector_type(16))) __bf16 v16bf;
typedef __attribute__((ext_vector_type(8)))  __bf16 v8bf;
typedef __attribute__((ext_vector_type(8)))  float  v8f;

static constexpr int M  = 8192;
static constexpr int K  = 4096;
static constexpr int N  = 12288;
static constexpr int BM = 128;
static constexpr int BN = 128;
static constexpr int BK = 32;          // == quant group size
static constexpr int LDST = 40;        // padded LDS row stride (80B): conflict-free frag reads
static constexpr int NK = K / BK;      // 128 k-steps

__global__ __launch_bounds__(256)
void int4gemm_wmma_bf16(const float* __restrict__ x,
                        const int*   __restrict__ w,
                        const float* __restrict__ sz,
                        const float* __restrict__ bias,
                        float* __restrict__ y)
{
    __shared__ __bf16 As[2][BM * LDST];   // activations, bf16
    __shared__ __bf16 Ws[2][BN * LDST];   // dequantized weights, bf16

    const int tid  = threadIdx.x;
    const int lane = tid & 31;
    const int wave = tid >> 5;
    const int lm   = lane & 15;   // row/col within 16x16 frag
    const int lh   = lane >> 4;   // K-half selector per ISA bf16 layout

    const int m0 = blockIdx.y * BM;
    const int n0 = blockIdx.x * BN;

    // Staging: 2 threads per row, 16 elements (4 x b128) each.
    const int sr = tid >> 1;            // 0..127
    const int sc = (tid & 1) << 4;      // 0 or 16

    const float* xg = x + (size_t)(m0 + sr) * K + sc;
    const int*   wg = w + (size_t)(n0 + sr) * K + sc;

    float4 fa[4];
    int4   qa[4];
    float  scale, zadj;

    auto load_regs = [&](int kt) {
        const float4* xp = (const float4*)(xg + (size_t)kt * BK);
        const int4*   wp = (const int4*)  (wg + (size_t)kt * BK);
#pragma unroll
        for (int i = 0; i < 4; ++i) fa[i] = xp[i];
#pragma unroll
        for (int i = 0; i < 4; ++i) qa[i] = wp[i];
        // group g == kt; [G][N][2] layout, 8B-aligned pair
        const float2 p = *(const float2*)(sz + ((size_t)kt * N + (n0 + sr)) * 2);
        scale = p.x;
        zadj  = p.y - 8.0f * p.x;   // fold the (q-8) shift into the zero point
    };

    auto store_lds = [&](int buf) {
        // activations fp32 -> bf16
        v8bf a0 = { (__bf16)fa[0].x, (__bf16)fa[0].y, (__bf16)fa[0].z, (__bf16)fa[0].w,
                    (__bf16)fa[1].x, (__bf16)fa[1].y, (__bf16)fa[1].z, (__bf16)fa[1].w };
        v8bf a1 = { (__bf16)fa[2].x, (__bf16)fa[2].y, (__bf16)fa[2].z, (__bf16)fa[2].w,
                    (__bf16)fa[3].x, (__bf16)fa[3].y, (__bf16)fa[3].z, (__bf16)fa[3].w };
        *(v8bf*)&As[buf][sr * LDST + sc]     = a0;
        *(v8bf*)&As[buf][sr * LDST + sc + 8] = a1;

        // int4 codes -> q*scale + zadj -> bf16  (no integer adds in the loop)
        v8bf w0 = { (__bf16)((float)qa[0].x * scale + zadj),
                    (__bf16)((float)qa[0].y * scale + zadj),
                    (__bf16)((float)qa[0].z * scale + zadj),
                    (__bf16)((float)qa[0].w * scale + zadj),
                    (__bf16)((float)qa[1].x * scale + zadj),
                    (__bf16)((float)qa[1].y * scale + zadj),
                    (__bf16)((float)qa[1].z * scale + zadj),
                    (__bf16)((float)qa[1].w * scale + zadj) };
        v8bf w1 = { (__bf16)((float)qa[2].x * scale + zadj),
                    (__bf16)((float)qa[2].y * scale + zadj),
                    (__bf16)((float)qa[2].z * scale + zadj),
                    (__bf16)((float)qa[2].w * scale + zadj),
                    (__bf16)((float)qa[3].x * scale + zadj),
                    (__bf16)((float)qa[3].y * scale + zadj),
                    (__bf16)((float)qa[3].z * scale + zadj),
                    (__bf16)((float)qa[3].w * scale + zadj) };
        *(v8bf*)&Ws[buf][sr * LDST + sc]     = w0;
        *(v8bf*)&Ws[buf][sr * LDST + sc + 8] = w1;
    };

    // wave tiling: 2x4 waves, each 64(M) x 32(N) = 4x2 fragments
    const int wm = (wave >> 2) * 64;
    const int wn = (wave & 3) * 32;

    v8f acc[4][2] = {};

    auto compute = [&](int buf) {
        v16bf a[4], b[2];
#pragma unroll
        for (int i = 0; i < 4; ++i) {
            const __bf16* p = &As[buf][(wm + i * 16 + lm) * LDST + lh * 8];
            v8bf lo = *(const v8bf*)p;
            v8bf hi = *(const v8bf*)(p + 16);
            a[i] = __builtin_shufflevector(lo, hi, 0,1,2,3,4,5,6,7,8,9,10,11,12,13,14,15);
        }
#pragma unroll
        for (int j = 0; j < 2; ++j) {
            const __bf16* p = &Ws[buf][(wn + j * 16 + lm) * LDST + lh * 8];
            v8bf lo = *(const v8bf*)p;
            v8bf hi = *(const v8bf*)(p + 16);
            b[j] = __builtin_shufflevector(lo, hi, 0,1,2,3,4,5,6,7,8,9,10,11,12,13,14,15);
        }
#pragma unroll
        for (int i = 0; i < 4; ++i)
#pragma unroll
            for (int j = 0; j < 2; ++j)
                acc[i][j] = __builtin_amdgcn_wmma_f32_16x16x32_bf16(
                    false, a[i], false, b[j], (short)0, acc[i][j], false, false);
    };

    // software pipeline: prologue fills buffer 0
    load_regs(0);
    store_lds(0);
    __syncthreads();

    for (int kt = 0; kt < NK; ++kt) {
        const int cur = kt & 1;
        if (kt + 1 < NK) load_regs(kt + 1);   // global loads for next tile overlap WMMA
        compute(cur);
        if (kt + 1 < NK) store_lds(cur ^ 1);
        __syncthreads();
    }

    // epilogue: C layout -> VGPR r holds row (r + 8*lh), lane&15 holds column
#pragma unroll
    for (int j = 0; j < 2; ++j) {
        const int col = n0 + wn + j * 16 + lm;
        const float bv = bias[col];
#pragma unroll
        for (int i = 0; i < 4; ++i) {
            const int rbase = m0 + wm + i * 16 + lh * 8;
#pragma unroll
            for (int r = 0; r < 8; ++r) {
                y[(size_t)(rbase + r) * N + col] = acc[i][j][r] + bv;
            }
        }
    }
}

extern "C" void kernel_launch(void* const* d_in, const int* in_sizes, int n_in,
                              void* d_out, int out_size, void* d_ws, size_t ws_size,
                              hipStream_t stream) {
    const float* x    = (const float*)d_in[0];   // [M,K] fp32
    const int*   wq   = (const int*)  d_in[1];   // [N,K] int codes
    const float* sz   = (const float*)d_in[2];   // [K/32,N,2] fp32
    const float* bias = (const float*)d_in[3];   // [N] fp32
    float*       y    = (float*)d_out;           // [M,N] fp32

    dim3 grid(N / BN, M / BM);   // 96 x 64 blocks
    int4gemm_wmma_bf16<<<grid, 256, 0, stream>>>(x, wq, sz, bias, y);
}
